// MyRNN_87591563035014
// MI455X (gfx1250) — matrix operand order repaired and checked by reading, not compiled, not run on hardware
//
#include <hip/hip_runtime.h>

typedef __attribute__((ext_vector_type(16))) _Float16 v16h;
typedef __attribute__((ext_vector_type(8)))  _Float16 v8h;
typedef __attribute__((ext_vector_type(8)))  float    v8f;

#define T_SEQ   80
#define EMBED   100
#define EPAD    128
#define UNITS   64
#define GATES   256
#define ROWS_WG 64
#define THREADS 256
#define BATCH   4096

// workspace (f16 elements): WMMA-B-swizzled weights
#define W1S_OFF 0          // 4 kt * 16 nt * 512 = 32768 (K padded 100->128; k=100 row = b1)
#define U1S_OFF 32768      // 2 kt * 16 nt * 512 = 16384
#define W2S_OFF 49152      // 16384
#define U2S_OFF 65536      // 16384
#define WS_TOTAL 81920     // f16 elems = 163840 bytes

// dynamic LDS layout:
// sW1 32768 | sU1 16384 | sW2 16384 | sU2 16384 | sX 8192 | sH1 4096 | sH2 4096  (f16)
// sH2f 4096 (f32)
#define SMEM_BYTES (98304*2 + 4096*4)   // 212992

// ---------------- prep: swizzle f32 weights into WMMA B-fragment order (f16) ----------
// B (KxN, K=32 per wmma step): lane n (0-15) holds col n, K {0..7, 16..23} in elems 0..15;
// lane 16+n holds col n, K {8..15, 24..31}.  Block (kt,nt) -> 32 lanes x 16 f16 contiguous.
// Layer-1 bias trick: W1 k==100 row carries b1 (matched by sX[:,100] == 1.0).
__global__ void prep_swizzle(const float* __restrict__ W1, const float* __restrict__ U1,
                             const float* __restrict__ W2, const float* __restrict__ U2,
                             const float* __restrict__ b1, _Float16* __restrict__ ws)
{
    int idx = blockIdx.x * blockDim.x + threadIdx.x;
    if (idx >= WS_TOTAL) return;
    const float* src; int Ksrc; int off; int isW1 = 0;
    if (idx < U1S_OFF)      { src = W1; Ksrc = 100; off = idx; isW1 = 1; }
    else if (idx < W2S_OFF) { src = U1; Ksrc = 64;  off = idx - U1S_OFF; }
    else if (idx < U2S_OFF) { src = W2; Ksrc = 64;  off = idx - W2S_OFF; }
    else                    { src = U2; Ksrc = 64;  off = idx - U2S_OFF; }
    int e    = off & 15;
    int lane = (off >> 4) & 31;
    int blk  = off >> 9;          // kt*16 + nt
    int nt   = blk & 15;
    int kt   = blk >> 4;
    int n = nt * 16 + (lane & 15);
    int k = kt * 32 + ((lane >> 4) << 3) + (e & 7) + ((e >> 3) << 4);
    float v;
    if (k < Ksrc)                 v = src[k * GATES + n];
    else if (isW1 && k == EMBED)  v = b1[n];          // bias row
    else                          v = 0.0f;
    ws[idx] = (_Float16)v;
}

// ---------------- fragment loaders ----------------
// A (16x32, 16-bit): lanes 0-15 row m, K {kb+0..7, kb+16..23}; lanes 16-31 K {kb+8..15, kb+24..31}
__device__ __forceinline__ v16h loadA(const _Float16* buf, int stride, int mt, int kt,
                                      int laneLo, int laneHi)
{
    const _Float16* p = buf + (mt * 16 + laneLo) * stride + kt * 32 + laneHi * 8;
    v8h lo = *(const v8h*)p;
    v8h hi = *(const v8h*)(p + 16);
    return __builtin_shufflevector(lo, hi, 0,1,2,3,4,5,6,7,8,9,10,11,12,13,14,15);
}

__device__ __forceinline__ v16h loadB(const _Float16* buf, int blk, int lane)
{
    return *(const v16h*)(buf + blk * 512 + lane * 16);
}

__device__ __forceinline__ v8f wmma16(v16h a, v16h b, v8f c)
{
    return __builtin_amdgcn_wmma_f32_16x16x32_f16(false, a, false, b, (short)0, c, false, false);
}

__device__ __forceinline__ float sigm(float x) { return 1.0f / (1.0f + __expf(-x)); }

__device__ __forceinline__ float tanh_hw(float x)
{
#if __has_builtin(__builtin_amdgcn_tanhf)
    return __builtin_amdgcn_tanhf(x);
#elif __has_builtin(__builtin_amdgcn_tanh_f32)
    return __builtin_amdgcn_tanh_f32(x);
#else
    float e = __expf(-2.0f * x);
    return (1.0f - e) / (1.0f + e);
#endif
}

// ---------------- main: full 80-step 2-layer LSTM per 64-row batch tile ----------------
__global__ __launch_bounds__(THREADS)
void lstm_main(const int* __restrict__ tokens, const float* __restrict__ emb,
               const float* __restrict__ b2, const float* __restrict__ Wo,
               const float* __restrict__ bo, const _Float16* __restrict__ wsw,
               float* __restrict__ out)
{
    extern __shared__ char smem_raw[] __attribute__((aligned(32)));
    _Float16* sW1 = (_Float16*)smem_raw;
    _Float16* sU1 = sW1 + 32768;
    _Float16* sW2 = sU1 + 16384;
    _Float16* sU2 = sW2 + 16384;
    _Float16* sX  = sU2 + 16384;    // 64 x 128 (col 100 = 1.0 bias lane; 101..127 zero)
    _Float16* sH1 = sX  + 8192;     // 64 x 64
    _Float16* sH2 = sH1 + 4096;     // 64 x 64
    float*    sH2f = (float*)(sH2 + 4096);   // 64 x 64 f32 (final h2)

    const int tid    = threadIdx.x;
    const int lane   = tid & 31;
    const int wave   = tid >> 5;
    const int mt     = wave & 3;
    const int utBase = (wave >> 2) << 1;   // waves 0-3: unit tiles {0,1}; 4-7: {2,3}
    const int laneLo = lane & 15;
    const int laneHi = lane >> 4;
    const int rowBase = blockIdx.x * ROWS_WG;
    const int gr = tid >> 2;               // gather: row within tile
    const int gq = tid & 3;                // gather: quarter of the 100 cols

    // layer-2 bias, resident in registers for all 80 steps
    float bias2[2][4];
    #pragma unroll
    for (int s = 0; s < 2; ++s)
        #pragma unroll
        for (int g = 0; g < 4; ++g)
            bias2[s][g] = b2[g * 64 + (utBase + s) * 16 + laneLo];

    // init: weights -> LDS, zero X/H1/H2, bias lane
    for (int i = tid; i < WS_TOTAL / 8; i += THREADS)
        ((v8h*)sW1)[i] = ((const v8h*)wsw)[i];
    {
        v8h z = {};
        for (int i = tid; i < 16384 / 8; i += THREADS)   // sX + sH1 + sH2 contiguous
            ((v8h*)sX)[i] = z;
    }

    v8f zv = {};
    v8f c1[2] = { zv, zv };
    v8f c2[2] = { zv, zv };

    // gather x(0)
    float xn[25];
    {
        int tok = tokens[(rowBase + gr) * T_SEQ + 0];
        const float* src = emb + tok * EMBED + gq * 25;
        #pragma unroll
        for (int i = 0; i < 25; ++i) xn[i] = src[i];
    }
    __syncthreads();
    if (tid < ROWS_WG) sX[tid * EPAD + EMBED] = (_Float16)1.0f;   // bias-lane "1"
    {
        _Float16* dst = sX + gr * EPAD + gq * 25;
        #pragma unroll
        for (int i = 0; i < 25; ++i) dst[i] = (_Float16)xn[i];
    }
    __syncthreads();

    #pragma unroll 1
    for (int t = 0; t < T_SEQ; ++t) {
        // issue gather for t+1 early; consumed at the bottom (overlaps WMMA)
        if (t + 1 < T_SEQ) {
            int tok = tokens[(rowBase + gr) * T_SEQ + (t + 1)];
            const float* src = emb + tok * EMBED + gq * 25;
            #pragma unroll
            for (int i = 0; i < 25; ++i) xn[i] = src[i];
        }

        v8f acc[2][4];

        // ---------------- layer 1: Z = [X,1]@[W1;b1] + H1@U1 (bias via K=100 row) ----
        #pragma unroll
        for (int s = 0; s < 2; ++s)
            #pragma unroll
            for (int g = 0; g < 4; ++g)
                acc[s][g] = zv;                         // inline-0 C on first WMMA
        #pragma unroll
        for (int kt = 0; kt < 4; ++kt) {
            v16h af = loadA(sX, EPAD, mt, kt, laneLo, laneHi);
            #pragma unroll
            for (int s = 0; s < 2; ++s) {
                int ut = utBase + s;
                #pragma unroll
                for (int g = 0; g < 4; ++g) {
                    v16h bf = loadB(sW1, kt * 16 + g * 4 + ut, lane);
                    acc[s][g] = wmma16(af, bf, acc[s][g]);
                }
            }
        }
        #pragma unroll
        for (int kt = 0; kt < 2; ++kt) {
            v16h af = loadA(sH1, UNITS, mt, kt, laneLo, laneHi);
            #pragma unroll
            for (int s = 0; s < 2; ++s) {
                int ut = utBase + s;
                #pragma unroll
                for (int g = 0; g < 4; ++g) {
                    v16h bf = loadB(sU1, kt * 16 + g * 4 + ut, lane);
                    acc[s][g] = wmma16(af, bf, acc[s][g]);
                }
            }
        }
        __syncthreads();   // all reads of sX / sH1(t-1) complete

        // gates layer 1 (in-register), write h1 -> LDS f16
        #pragma unroll
        for (int s = 0; s < 2; ++s) {
            int ut = utBase + s;
            #pragma unroll
            for (int v = 0; v < 8; ++v) {
                float ig = sigm(acc[s][0][v]);
                float fg = sigm(acc[s][1][v]);
                float gg = tanh_hw(acc[s][2][v]);
                float og = sigm(acc[s][3][v]);
                float c  = fg * c1[s][v] + ig * gg;
                c1[s][v] = c;
                float h  = og * tanh_hw(c);
                int r = mt * 16 + (laneHi << 3) + v;
                int u = ut * 16 + laneLo;
                sH1[r * UNITS + u] = (_Float16)h;
            }
        }
        __syncthreads();   // h1(t) visible

        // ---------------- layer 2: Z = H1@W2 + H2@U2 + b2 ----------------
        #pragma unroll
        for (int s = 0; s < 2; ++s)
            #pragma unroll
            for (int g = 0; g < 4; ++g)
                acc[s][g] = zv;                         // inline-0 C on first WMMA
        #pragma unroll
        for (int kt = 0; kt < 2; ++kt) {
            v16h af = loadA(sH1, UNITS, mt, kt, laneLo, laneHi);
            #pragma unroll
            for (int s = 0; s < 2; ++s) {
                int ut = utBase + s;
                #pragma unroll
                for (int g = 0; g < 4; ++g) {
                    v16h bf = loadB(sW2, kt * 16 + g * 4 + ut, lane);
                    acc[s][g] = wmma16(af, bf, acc[s][g]);
                }
            }
        }
        #pragma unroll
        for (int kt = 0; kt < 2; ++kt) {
            v16h af = loadA(sH2, UNITS, mt, kt, laneLo, laneHi);
            #pragma unroll
            for (int s = 0; s < 2; ++s) {
                int ut = utBase + s;
                #pragma unroll
                for (int g = 0; g < 4; ++g) {
                    v16h bf = loadB(sU2, kt * 16 + g * 4 + ut, lane);
                    acc[s][g] = wmma16(af, bf, acc[s][g]);
                }
            }
        }
        __syncthreads();   // all reads of sH2(t-1) complete

        // gates layer 2 (bias from registers), write h2 -> LDS f16 (+f32 at final step)
        #pragma unroll
        for (int s = 0; s < 2; ++s) {
            int ut = utBase + s;
            #pragma unroll
            for (int v = 0; v < 8; ++v) {
                float ig = sigm(acc[s][0][v] + bias2[s][0]);
                float fg = sigm(acc[s][1][v] + bias2[s][1]);
                float gg = tanh_hw(acc[s][2][v] + bias2[s][2]);
                float og = sigm(acc[s][3][v] + bias2[s][3]);
                float c  = fg * c2[s][v] + ig * gg;
                c2[s][v] = c;
                float h  = og * tanh_hw(c);
                int r = mt * 16 + (laneHi << 3) + v;
                int u = ut * 16 + laneLo;
                sH2[r * UNITS + u] = (_Float16)h;
                if (t == T_SEQ - 1) sH2f[r * UNITS + u] = h;
            }
        }

        // commit x(t+1) to LDS (waits on the early gather)
        if (t + 1 < T_SEQ) {
            _Float16* dst = sX + gr * EPAD + gq * 25;
            #pragma unroll
            for (int i = 0; i < 25; ++i) dst[i] = (_Float16)xn[i];
        }
        __syncthreads();
    }

    // output: sigmoid(h2 @ Wo + bo)
    if (tid < ROWS_WG) {
        float s = 0.0f;
        #pragma unroll
        for (int u = 0; u < UNITS; ++u) s += sH2f[tid * UNITS + u] * Wo[u];
        s += bo[0];
        out[rowBase + tid] = 1.0f / (1.0f + __expf(-s));
    }
}

extern "C" void kernel_launch(void* const* d_in, const int* in_sizes, int n_in,
                              void* d_out, int out_size, void* d_ws, size_t ws_size,
                              hipStream_t stream)
{
    const int*   tokens = (const int*)  d_in[0];
    const float* emb    = (const float*)d_in[1];
    const float* W1     = (const float*)d_in[2];
    const float* U1     = (const float*)d_in[3];
    const float* b1     = (const float*)d_in[4];
    const float* W2     = (const float*)d_in[5];
    const float* U2     = (const float*)d_in[6];
    const float* b2     = (const float*)d_in[7];
    const float* Wo     = (const float*)d_in[8];
    const float* bo     = (const float*)d_in[9];
    _Float16* wsw = (_Float16*)d_ws;
    float* out = (float*)d_out;

    prep_swizzle<<<(WS_TOTAL + THREADS - 1) / THREADS, THREADS, 0, stream>>>(W1, U1, W2, U2,
                                                                            b1, wsw);

    (void)hipFuncSetAttribute(reinterpret_cast<const void*>(lstm_main),
                              hipFuncAttributeMaxDynamicSharedMemorySize, (int)SMEM_BYTES);
    lstm_main<<<BATCH / ROWS_WG, THREADS, SMEM_BYTES, stream>>>(tokens, emb, b2, Wo, bo,
                                                               wsw, out);
}
